// DecoderBlock_82514911690872
// MI455X (gfx1250) — compile-verified, matrix operand order checked by reading, not run on hardware
//
#include <hip/hip_runtime.h>

// ---------------------------------------------------------------------------
// Decoder block for MI455X (gfx1250): bf16 WMMA everywhere, f32 accumulate.
// Async LDS staging (GLOBAL_LOAD_ASYNC_TO_LDS_B128, ASYNCcnt) + double-buffered
// GEMM mainloop.  S=4096, D=1024, H=16, dh=64, Dff=4096.
// ---------------------------------------------------------------------------

typedef __bf16 bf16_t;
typedef __attribute__((ext_vector_type(16))) __bf16 v16bf;
typedef __attribute__((ext_vector_type(8)))  __bf16 v8bf;
typedef __attribute__((ext_vector_type(8)))  float  v8f;

constexpr int S_LEN   = 4096;
constexpr int D_MODEL = 1024;
constexpr int N_HEADS = 16;
constexpr int D_HEAD  = 64;
constexpr int D_FF    = 4096;

// ---- CDNA5 async copy: global -> LDS, 16B per lane, tracked by ASYNCcnt ----
__device__ __forceinline__ void async_copy_b128(void* lds, const void* gptr) {
  // Low 32 bits of a generic pointer into the LDS aperture are the
  // workgroup-relative LDS byte offset (ISA 10.2): exactly what VDST wants.
  unsigned l = (unsigned)(uintptr_t)lds;
  asm volatile("global_load_async_to_lds_b128 %0, %1, off"
               :: "v"(l), "v"(gptr)
               : "memory");
}
__device__ __forceinline__ void wait_async() {
  asm volatile("s_wait_asynccnt 0x0" ::: "memory");
}

// ---- WMMA helpers ----------------------------------------------------------
// CDNA5 16-bit A-matrix 16x32 layout (ISA 7.12.2): lane L holds row M=L&15;
// VGPRs 0..3 hold K = half*8 .. +7, VGPRs 4..7 hold K = 16+half*8 .. +7,
// half = L>>4.  Two 16-byte loads per lane.  B fragments (B stored N-major
// [N][K]) use the identical loader.
__device__ __forceinline__ v16bf load_frag(const bf16_t* base, int ld) {
  const int lane = threadIdx.x & 31;
  const int half = lane >> 4;
  const bf16_t* row = base + (lane & 15) * ld + half * 8;
  v8bf lo = *(const v8bf*)(row);
  v8bf hi = *(const v8bf*)(row + 16);
  v16bf f;
#pragma unroll
  for (int i = 0; i < 8; ++i) { f[i] = lo[i]; f[i + 8] = hi[i]; }
  return f;
}

__device__ __forceinline__ v8f wmma_bf16(v16bf a, v16bf b, v8f c) {
  // (neg_a, A, neg_b, B, c_mod, C, reuse_a, reuse_b)
  return __builtin_amdgcn_wmma_f32_16x16x32_bf16(false, a, false, b,
                                                 (short)0, c, false, false);
}

// ---- weight conversion -----------------------------------------------------
// Fused QKV weight, N-major: Bt[j][d], j in [0,3D): j<D -> Q, <2D -> K, else V.
__global__ __launch_bounds__(256) void conv_qkv_kernel(
    const float* __restrict__ WQ, const float* __restrict__ WK,
    const float* __restrict__ WV, bf16_t* __restrict__ Bt) {
  int idx = blockIdx.x * 256 + threadIdx.x;          // over 3*D*D
  if (idx >= 3 * D_MODEL * D_MODEL) return;
  int j  = idx >> 10;                                 // / D_MODEL
  int d  = idx & (D_MODEL - 1);
  int sec = j >> 10;
  int jj  = j & (D_MODEL - 1);
  int hh = jj >> 6, e = jj & 63;                      // head, dh index
  const float* W = (sec == 0) ? WQ : (sec == 1) ? WK : WV;
  Bt[idx] = (bf16_t)W[((size_t)hh * D_MODEL + d) * D_HEAD + e];
}

// Generic transpose+convert: W is [K][N] f32 row-major -> Bt [N][K] bf16.
__global__ __launch_bounds__(256) void tconv_kernel(
    const float* __restrict__ W, bf16_t* __restrict__ Bt, int K, int N) {
  int idx = blockIdx.x * 256 + threadIdx.x;           // over N*K
  if (idx >= N * K) return;
  int n = idx / K, d = idx - n * K;
  Bt[idx] = (bf16_t)W[(size_t)d * N + n];
}

// ---- RMSNorm -> bf16 -------------------------------------------------------
__global__ __launch_bounds__(256) void rmsnorm_kernel(
    const float* __restrict__ x, const float* __restrict__ g,
    bf16_t* __restrict__ out) {
  const int row = blockIdx.x;
  const int tid = threadIdx.x;
  const float* xr = x + (size_t)row * D_MODEL;
  float ss = 0.f;
#pragma unroll
  for (int i = 0; i < D_MODEL / 256; ++i) {
    float v = xr[tid + i * 256];
    ss += v * v;
  }
#pragma unroll
  for (int off = 16; off > 0; off >>= 1) ss += __shfl_xor(ss, off, 32);
  __shared__ float red[8];
  if ((tid & 31) == 0) red[tid >> 5] = ss;
  __syncthreads();
  float tot = 0.f;
#pragma unroll
  for (int i = 0; i < 8; ++i) tot += red[i];
  float inv = rsqrtf(tot * (1.0f / D_MODEL) + 1e-6f);
#pragma unroll
  for (int i = 0; i < D_MODEL / 256; ++i) {
    int c = tid + i * 256;
    out[(size_t)row * D_MODEL + c] = (bf16_t)(xr[c] * inv * g[c]);
  }
}

// ---- tiled WMMA GEMM, double-buffered async staging ------------------------
// C[M,N] = A[M,K] (bf16, row-major) x Bt[N,K] (bf16, N-major).
// Block tile 128x128, BK=32, 8 waves in 2(M)x4(N), 64x32 per wave
// (4 M-frags x 2 N-frags -> 8 wmma per k-step per wave).
// EPI: 0 = QKV scatter (Q,K [H][S][dh], V transposed [H][dh][S], all bf16)
//      1 = f32 out, optional bias + residual
//      2 = bias + ReLU -> bf16 out
template <int EPI>
__global__ __launch_bounds__(256) void gemm_bf16_kernel(
    const bf16_t* __restrict__ A, const bf16_t* __restrict__ Bt,
    const float* __restrict__ bias, const float* __restrict__ resid,
    float* __restrict__ outf, bf16_t* __restrict__ outb,
    bf16_t* __restrict__ q_out, bf16_t* __restrict__ k_out,
    bf16_t* __restrict__ vt_out, int M, int N, int K) {
  constexpr int BM = 128, BN = 128, BK = 32, LDA = BK + 8;  // 80B stride
  __shared__ bf16_t As[2][BM][LDA];
  __shared__ bf16_t Bs[2][BN][LDA];
  const int tid  = threadIdx.x;
  const int wid  = tid >> 5;
  const int lane = tid & 31;
  const int half = lane >> 4;
  const int nl   = lane & 15;
  const int wm = wid >> 2, wn = wid & 3;
  const int m0 = blockIdx.y * BM, n0 = blockIdx.x * BN;

  v8f acc[4][2] = {};

  auto stage = [&](int buf, int k0) {
#pragma unroll
    for (int i = 0; i < 2; ++i) {                      // 512 16B segments
      int seg = tid + i * 256;
      int r = seg >> 2, cs = (seg & 3) * 8;
      async_copy_b128(&As[buf][r][cs], A  + (size_t)(m0 + r) * K + k0 + cs);
      async_copy_b128(&Bs[buf][r][cs], Bt + (size_t)(n0 + r) * K + k0 + cs);
    }
  };

  stage(0, 0);
  wait_async();
  __syncthreads();

  int buf = 0;
  for (int k0 = 0; k0 < K; k0 += BK) {
    if (k0 + BK < K) stage(buf ^ 1, k0 + BK);   // prefetch next tile (async)

    v16bf af[4], bfr[2];
#pragma unroll
    for (int mf = 0; mf < 4; ++mf)
      af[mf]  = load_frag(&As[buf][wm * 64 + mf * 16][0], LDA);
#pragma unroll
    for (int nf = 0; nf < 2; ++nf)
      bfr[nf] = load_frag(&Bs[buf][wn * 32 + nf * 16][0], LDA);
#pragma unroll
    for (int mf = 0; mf < 4; ++mf)
#pragma unroll
      for (int nf = 0; nf < 2; ++nf)
        acc[mf][nf] = wmma_bf16(af[mf], bfr[nf], acc[mf][nf]);

    wait_async();
    __syncthreads();
    buf ^= 1;
  }

  // Epilogue.  C/D layout: lane holds col n = lane&15, rows v + 8*(lane>>4).
#pragma unroll
  for (int mf = 0; mf < 4; ++mf)
#pragma unroll
    for (int nf = 0; nf < 2; ++nf)
#pragma unroll
      for (int v = 0; v < 8; ++v) {
        int row = m0 + wm * 64 + mf * 16 + v + 8 * half;
        int col = n0 + wn * 32 + nf * 16 + nl;
        float c = acc[mf][nf][v];
        if (EPI == 0) {
          int sec = col >> 10;                  // 0:Q 1:K 2:V
          int jj  = col & (D_MODEL - 1);
          int hh = jj >> 6, e = jj & 63;
          if (sec == 0)
            q_out[((size_t)hh * S_LEN + row) * D_HEAD + e] = (bf16_t)c;
          else if (sec == 1)
            k_out[((size_t)hh * S_LEN + row) * D_HEAD + e] = (bf16_t)c;
          else
            vt_out[((size_t)hh * D_HEAD + e) * S_LEN + row] = (bf16_t)c;
        } else if (EPI == 1) {
          float r = c;
          if (bias)  r += bias[col];
          if (resid) r += resid[(size_t)row * N + col];
          outf[(size_t)row * N + col] = r;
        } else {
          float r = c + bias[col];
          r = r > 0.f ? r : 0.f;
          outb[(size_t)row * N + col] = (bf16_t)r;
        }
      }
}

// ---- flash attention -------------------------------------------------------
// Q,K bf16 [H][S][dh]; Vt bf16 [H][dh][S]; O bf16 [S][D] (heads concatenated).
// Block: 128 queries x 1 head, 8 waves x 16 query rows.  Key blocks of 64,
// causal.  Online softmax on WMMA f32 accumulators; P re-fragmented via LDS.
__global__ __launch_bounds__(256) void attn_kernel(
    const bf16_t* __restrict__ Qm, const bf16_t* __restrict__ Km,
    const bf16_t* __restrict__ Vt, bf16_t* __restrict__ O) {
  __shared__ bf16_t Ks[64][72];       // [key][dh]
  __shared__ bf16_t Vs[64][72];       // [dh][key]  (V transposed)
  __shared__ bf16_t Ps[8][16][72];    // per-wave P tile [q][key]
  const int tid = threadIdx.x, wid = tid >> 5;
  const int lane = tid & 31, half = lane >> 4, nl = lane & 15;
  const int h = blockIdx.y;
  const int qbase = blockIdx.x * 128;
  const int q0 = qbase + wid * 16;

  const bf16_t* qptr = Qm + ((size_t)h * S_LEN + q0) * D_HEAD;
  v16bf aq[2];
  aq[0] = load_frag(qptr, D_HEAD);
  aq[1] = load_frag(qptr + 32, D_HEAD);

  float mrow[8], lrow[8];
#pragma unroll
  for (int v = 0; v < 8; ++v) { mrow[v] = -1e30f; lrow[v] = 0.f; }
  v8f oacc[4] = {};

  const int nkb = qbase / 64 + 2;     // causal extent for this query block
  for (int jb = 0; jb < nkb; ++jb) {
    __syncthreads();                  // everyone done reading Ks/Vs/Ps
#pragma unroll
    for (int i = 0; i < 2; ++i) {                      // 512 16B segments
      int seg = tid + i * 256;
      int r = seg >> 3, cs = (seg & 7) * 8;
      async_copy_b128(&Ks[r][cs],
          Km + ((size_t)h * S_LEN + jb * 64 + r) * D_HEAD + cs);
      async_copy_b128(&Vs[r][cs],
          Vt + ((size_t)h * D_HEAD + r) * S_LEN + jb * 64 + cs);
    }
    wait_async();
    __syncthreads();

    // S = Q K^T  (16 q rows x 64 keys per wave)
    v8f sc[4] = {};
#pragma unroll
    for (int nf = 0; nf < 4; ++nf)
#pragma unroll
      for (int kf = 0; kf < 2; ++kf) {
        v16bf bk = load_frag(&Ks[nf * 16][kf * 32], 72);
        sc[nf] = wmma_bf16(aq[kf], bk, sc[nf]);
      }

    // scale + causal mask
#pragma unroll
    for (int nf = 0; nf < 4; ++nf)
#pragma unroll
      for (int v = 0; v < 8; ++v) {
        int key = jb * 64 + nf * 16 + nl;
        int qr  = q0 + v + 8 * half;
        float s = sc[nf][v] * 0.125f;                  // 1/sqrt(64)
        sc[nf][v] = (key > qr) ? -1e30f : s;
      }

    // online softmax: row stats replicated across the 16 lanes of a half
    float corr[8];
#pragma unroll
    for (int v = 0; v < 8; ++v) {
      float t = fmaxf(fmaxf(sc[0][v], sc[1][v]), fmaxf(sc[2][v], sc[3][v]));
#pragma unroll
      for (int off = 8; off > 0; off >>= 1) t = fmaxf(t, __shfl_xor(t, off, 16));
      float mn = fmaxf(mrow[v], t);
      corr[v] = __expf(mrow[v] - mn);
      mrow[v] = mn;
      lrow[v] *= corr[v];
    }
#pragma unroll
    for (int nf = 0; nf < 4; ++nf)
#pragma unroll
      for (int v = 0; v < 8; ++v) oacc[nf][v] *= corr[v];

    float psum[8] = {};
#pragma unroll
    for (int nf = 0; nf < 4; ++nf)
#pragma unroll
      for (int v = 0; v < 8; ++v) {
        float p = __expf(sc[nf][v] - mrow[v]);
        Ps[wid][v + 8 * half][nf * 16 + nl] = (bf16_t)p;
        psum[v] += p;
      }
#pragma unroll
    for (int v = 0; v < 8; ++v) {
      float t = psum[v];
#pragma unroll
      for (int off = 8; off > 0; off >>= 1) t += __shfl_xor(t, off, 16);
      lrow[v] += t;
    }
    __syncthreads();

    // O += P V   (B operand = Vs, dh-major)
#pragma unroll
    for (int kf = 0; kf < 2; ++kf) {
      v16bf ap = load_frag(&Ps[wid][0][kf * 32], 72);
#pragma unroll
      for (int nf = 0; nf < 4; ++nf) {
        v16bf bv = load_frag(&Vs[nf * 16][kf * 32], 72);
        oacc[nf] = wmma_bf16(ap, bv, oacc[nf]);
      }
    }
  }

  // normalize + store heads-concatenated bf16
#pragma unroll
  for (int nf = 0; nf < 4; ++nf)
#pragma unroll
    for (int v = 0; v < 8; ++v) {
      int row = q0 + v + 8 * half;
      int col = h * D_HEAD + nf * 16 + nl;
      O[(size_t)row * D_MODEL + col] = (bf16_t)(oacc[nf][v] / lrow[v]);
    }
}

// ---------------------------------------------------------------------------
extern "C" void kernel_launch(void* const* d_in, const int* in_sizes, int n_in,
                              void* d_out, int out_size, void* d_ws,
                              size_t ws_size, hipStream_t stream) {
  (void)in_sizes; (void)n_in; (void)out_size; (void)ws_size;
  const float* x  = (const float*)d_in[0];
  /* d_in[1] = mask: causal, computed arithmetically instead */
  const float* g1 = (const float*)d_in[2];
  const float* WQ = (const float*)d_in[3];
  const float* WK = (const float*)d_in[4];
  const float* WV = (const float*)d_in[5];
  const float* WO = (const float*)d_in[6];
  const float* g2 = (const float*)d_in[7];
  const float* W1 = (const float*)d_in[8];
  const float* B1 = (const float*)d_in[9];
  const float* W2 = (const float*)d_in[10];
  const float* B2 = (const float*)d_in[11];
  float* out = (float*)d_out;

  char* ws = (char*)d_ws;
  size_t off = 0;
  auto alloc = [&](size_t bytes) -> void* {
    void* p = ws + off;
    off += (bytes + 255) & ~(size_t)255;
    return p;
  };
  const size_t SD = (size_t)S_LEN * D_MODEL;
  bf16_t* wqkv_t = (bf16_t*)alloc((size_t)3 * D_MODEL * D_MODEL * 2);
  bf16_t* wo_t   = (bf16_t*)alloc((size_t)D_MODEL * D_MODEL * 2);
  bf16_t* w1_t   = (bf16_t*)alloc((size_t)D_FF * D_MODEL * 2);
  bf16_t* w2_t   = (bf16_t*)alloc((size_t)D_MODEL * D_FF * 2);
  bf16_t* hb     = (bf16_t*)alloc(SD * 2);           // rmsnorm(x,g1) bf16
  bf16_t* qb     = (bf16_t*)alloc(SD * 2);           // [H][S][dh]
  bf16_t* kb     = (bf16_t*)alloc(SD * 2);           // [H][S][dh]
  bf16_t* vtb    = (bf16_t*)alloc(SD * 2);           // [H][dh][S]
  bf16_t* ob     = (bf16_t*)alloc(SD * 2);           // attn out [S][D]
  float*  x1     = (float*)alloc(SD * 4);            // x + o@WO
  bf16_t* h2b    = (bf16_t*)alloc(SD * 2);           // rmsnorm(x1,g2) bf16
  bf16_t* m1b    = (bf16_t*)alloc((size_t)S_LEN * D_FF * 2);  // relu(h2@W1+B1)

  // 1. weight conversion (f32 -> bf16, N-major)
  conv_qkv_kernel<<<(3 * D_MODEL * D_MODEL + 255) / 256, 256, 0, stream>>>(
      WQ, WK, WV, wqkv_t);
  tconv_kernel<<<(D_MODEL * D_MODEL + 255) / 256, 256, 0, stream>>>(
      WO, wo_t, D_MODEL, D_MODEL);
  tconv_kernel<<<(D_MODEL * D_FF + 255) / 256, 256, 0, stream>>>(
      W1, w1_t, D_MODEL, D_FF);
  tconv_kernel<<<(D_FF * D_MODEL + 255) / 256, 256, 0, stream>>>(
      W2, w2_t, D_FF, D_MODEL);

  // 2. h = rmsnorm(x, g1)
  rmsnorm_kernel<<<S_LEN, 256, 0, stream>>>(x, g1, hb);

  // 3. fused QKV projection (M=S, N=3D, K=D), scatter Q/K/V^T
  gemm_bf16_kernel<0><<<dim3(3 * D_MODEL / 128, S_LEN / 128), 256, 0, stream>>>(
      hb, wqkv_t, nullptr, nullptr, nullptr, nullptr, qb, kb, vtb,
      S_LEN, 3 * D_MODEL, D_MODEL);

  // 4. causal flash attention -> ob [S][D]
  attn_kernel<<<dim3(S_LEN / 128, N_HEADS), 256, 0, stream>>>(qb, kb, vtb, ob);

  // 5. x1 = x + ob @ WO
  gemm_bf16_kernel<1><<<dim3(D_MODEL / 128, S_LEN / 128), 256, 0, stream>>>(
      ob, wo_t, nullptr, x, x1, nullptr, nullptr, nullptr, nullptr,
      S_LEN, D_MODEL, D_MODEL);

  // 6. h2 = rmsnorm(x1, g2)
  rmsnorm_kernel<<<S_LEN, 256, 0, stream>>>(x1, g2, h2b);

  // 7. m1 = relu(h2 @ W1 + B1), bf16
  gemm_bf16_kernel<2><<<dim3(D_FF / 128, S_LEN / 128), 256, 0, stream>>>(
      h2b, w1_t, B1, nullptr, nullptr, m1b, nullptr, nullptr, nullptr,
      S_LEN, D_FF, D_MODEL);

  // 8. out = x1 + m1 @ W2 + B2
  gemm_bf16_kernel<1><<<dim3(D_MODEL / 128, S_LEN / 128), 256, 0, stream>>>(
      m1b, w2_t, B2, x1, out, nullptr, nullptr, nullptr, nullptr,
      S_LEN, D_MODEL, D_FF);
}
